// ShapletLearner_54717883351092
// MI455X (gfx1250) — matrix operand order repaired and checked by reading, not compiled, not run on hardware
//
#include <hip/hip_runtime.h>

#define B_     4096
#define Q_     1024
#define L_     64
#define K_     32
#define S_     993        // Q - K + 1
#define STILES 63         // ceil(S/16), 63*16 = 1008
#define NT     4          // L / 16
#define NWAVES 8
#define BIGF   1e30f

typedef __attribute__((ext_vector_type(16))) __bf16 v16bf;
typedef __attribute__((ext_vector_type(8)))  float  v8f;

__launch_bounds__(256, 2)
__global__ void shapelet_min_dist_kernel(const float* __restrict__ ts,
                                         const float* __restrict__ sh,
                                         const float* __restrict__ fcw,
                                         const float* __restrict__ fcb,
                                         float* __restrict__ out)
{
    __shared__ float  ts_f[Q_];
    __shared__ __bf16 ts_hi[Q_ + 16];
    __shared__ __bf16 ts_lo[Q_ + 16];
    __shared__ __bf16 sh_hi[L_ * K_];
    __shared__ __bf16 sh_lo[L_ * K_];
    __shared__ float  ssq[L_];
    __shared__ float  wsq[STILES * 16];
    __shared__ float  featbuf[NWAVES][L_];

    const int tid  = threadIdx.x;
    const int b    = blockIdx.x;
    const int lane = tid & 31;
    const int wave = tid >> 5;

    // ---- stage time-series row: f32 + bf16 hi/lo split ----
    for (int i = tid; i < Q_; i += 256) {
        float  x = ts[b * Q_ + i];
        __bf16 h = (__bf16)x;
        ts_f[i]  = x;
        ts_hi[i] = h;
        ts_lo[i] = (__bf16)(x - (float)h);
    }
    if (tid < 16) {
        ts_hi[Q_ + tid] = (__bf16)0.0f;   // pad: last S-tile reads zeros (finite)
        ts_lo[Q_ + tid] = (__bf16)0.0f;
    }

    // ---- stage shapelets: bf16 hi/lo split + row norms ----
    for (int i = tid; i < L_ * K_; i += 256) {
        float  x = sh[i];
        __bf16 h = (__bf16)x;
        sh_hi[i] = h;
        sh_lo[i] = (__bf16)(x - (float)h);
    }
    if (tid < L_) {
        float acc = 0.0f;
        for (int k = 0; k < K_; ++k) { float v = sh[tid * K_ + k]; acc += v * v; }
        ssq[tid] = acc;
    }
    __syncthreads();

    // ---- sliding-window sum of squares; pad rows get +BIG so the min
    //      masks them with zero inner-loop cost (fully branchless) ----
    for (int s = tid; s < STILES * 16; s += 256) {
        float acc = BIGF;
        if (s < S_) {
            acc = 0.0f;
            for (int k = 0; k < K_; ++k) { float v = ts_f[s + k]; acc += v * v; }
        }
        wsq[s] = acc;
    }
    __syncthreads();

    // ---- B fragments: shapeletsT, resident in VGPRs for the whole loop ----
    // B 32x16, contiguous k order used here: lanes 0-15 hold K=0..15, lanes 16-31 hold K=16..31,
    // column N = lane & 15, vector element i <-> K = kbase + i.
    const int n   = lane & 15;
    const int kbB = (lane < 16) ? 0 : 16;
    v16bf bh[NT], bl[NT];
    float ssq_n[NT];
#pragma unroll
    for (int nt = 0; nt < NT; ++nt) {
        const int l = nt * 16 + n;
#pragma unroll
        for (int i = 0; i < 16; ++i) {
            bh[nt][i] = sh_hi[l * K_ + kbB + i];
            bl[nt][i] = sh_lo[l * K_ + kbB + i];
        }
        ssq_n[nt] = ssq[l];
    }

    // A 16x32, same contiguous k order as B: row M = lane&15; lane<16 -> K in {0..15},
    // lane>=16 -> K in {16..31}; element i -> K = kbA + i.
    const int kbA = (lane < 16) ? 0 : 16;
    const int mA  = lane & 15;
    const int mC  = (lane < 16) ? 0 : 8;   // C/D: VGPR r holds row M = r + mC

    // track min_s (wsq - 2*cross); (+ssq)/K applied once after the loop
    float mn[NT];
#pragma unroll
    for (int nt = 0; nt < NT; ++nt) mn[nt] = BIGF;

    // ---- main loop: each wave owns S-tiles wave, wave+8, ... ----
    for (int t = wave; t < STILES; t += NWAVES) {
        const int s0 = t * 16;

        v16bf ah, al;
#pragma unroll
        for (int i = 0; i < 16; ++i) {
            const int k = kbA + i;
            ah[i] = ts_hi[s0 + mA + k];
            al[i] = ts_lo[s0 + mA + k];
        }

        float wr[8];
#pragma unroll
        for (int r = 0; r < 8; ++r) wr[r] = wsq[s0 + mC + r];

#pragma unroll
        for (int nt = 0; nt < NT; ++nt) {
            v8f c = {};
            // cross = hi*hi + hi*lo + lo*hi  (~f32-accurate bf16 split)
            c = __builtin_amdgcn_wmma_f32_16x16x32_bf16(false, ah, false, bh[nt],
                                                        (short)0, c, false, false);
            c = __builtin_amdgcn_wmma_f32_16x16x32_bf16(false, ah, false, bl[nt],
                                                        (short)0, c, false, false);
            c = __builtin_amdgcn_wmma_f32_16x16x32_bf16(false, al, false, bh[nt],
                                                        (short)0, c, false, false);
#pragma unroll
            for (int r = 0; r < 8; ++r)
                mn[nt] = fminf(mn[nt], fmaf(-2.0f, c[r], wr[r]));
        }
    }

    // ---- finish distance transform, reduce the two lane halves ----
#pragma unroll
    for (int nt = 0; nt < NT; ++nt) {
        float v = (mn[nt] + ssq_n[nt]) * (1.0f / K_);
        float other = __shfl_xor(v, 16, 32);
        mn[nt] = fminf(v, other);
    }
    if (lane < 16) {
#pragma unroll
        for (int nt = 0; nt < NT; ++nt)
            featbuf[wave][nt * 16 + lane] = mn[nt];
    }
    __syncthreads();

    // ---- cross-wave min, then the tiny 64->2 FC ----
    if (tid < L_) {
        float f = featbuf[0][tid];
        for (int w = 1; w < NWAVES; ++w) f = fminf(f, featbuf[w][tid]);
        featbuf[0][tid] = f;
    }
    __syncthreads();
    if (tid < 2) {
        float acc = fcb[tid];
        for (int l = 0; l < L_; ++l) acc += featbuf[0][l] * fcw[tid * L_ + l];
        out[b * 2 + tid] = acc;
    }
}

extern "C" void kernel_launch(void* const* d_in, const int* in_sizes, int n_in,
                              void* d_out, int out_size, void* d_ws, size_t ws_size,
                              hipStream_t stream) {
    (void)in_sizes; (void)n_in; (void)out_size; (void)d_ws; (void)ws_size;
    const float* ts  = (const float*)d_in[0];
    const float* sh  = (const float*)d_in[1];
    const float* fcw = (const float*)d_in[2];
    const float* fcb = (const float*)d_in[3];
    float* out = (float*)d_out;
    shapelet_min_dist_kernel<<<B_, 256, 0, stream>>>(ts, sh, fcw, fcb, out);
}
